// MyLstmModel_71949292142775
// MI455X (gfx1250) — matrix operand order repaired and checked by reading, not compiled, not run on hardware
//
#include <hip/hip_runtime.h>
#include <cmath>

// ---------------------------------------------------------------------------
// CDNA5 (gfx1250) fused bidirectional LSTM stack, v3.1.
//  - B (weight) WMMA fragments register-resident across the whole T loop
//  - direction templated: no runtime selects on the per-step scalar path
//  - 2 barriers/step; x_{t+1} staged via GLOBAL_LOAD_ASYNC_TO_LDS_B32 into a
//    double-buffered LDS tile (ASYNCcnt) for bf16 layers; register pipeline
//    with clamped (branch-free) loads for the f32 input layer
//  - v_tanh_f32 / v_exp_f32 gate math (TRANS ops co-issue with XDL WMMA)
// ---------------------------------------------------------------------------

typedef __bf16 bf16;
typedef bf16 v16bf __attribute__((ext_vector_type(16)));
typedef float v8f __attribute__((ext_vector_type(8)));
typedef unsigned int v8u __attribute__((ext_vector_type(8)));

#if __has_builtin(__builtin_amdgcn_global_load_async_to_lds_b32) && \
    __has_builtin(__builtin_amdgcn_s_wait_asynccnt)
#define ASYNC_LDS_OK 1
#else
#define ASYNC_LDS_OK 0
#endif

#if ASYNC_LDS_OK
typedef __attribute__((address_space(1))) int gas_int;  // global addrspace
typedef __attribute__((address_space(3))) int las_int;  // LDS addrspace
__device__ __forceinline__ void async_g2l_b32(const void* g, void* l) {
  __builtin_amdgcn_global_load_async_to_lds_b32((gas_int*)g, (las_int*)l, 0, 0);
}
__device__ __forceinline__ void wait_async0() { __builtin_amdgcn_s_wait_asynccnt(0); }
#endif

__device__ __forceinline__ unsigned short f2bf(float f) {
  unsigned int u = __float_as_uint(f);
  u += 0x7FFFu + ((u >> 16) & 1u);  // round-to-nearest-even
  return (unsigned short)(u >> 16);
}
#if __has_builtin(__builtin_amdgcn_tanhf)
__device__ __forceinline__ float ftanh(float x) { return __builtin_amdgcn_tanhf(x); }
#else
__device__ __forceinline__ float ftanh(float x) { return tanhf(x); }
#endif
__device__ __forceinline__ float fsigm(float x) {
  return __builtin_amdgcn_rcpf(1.0f + __builtin_amdgcn_exp2f(-1.442695041f * x));
}

// D: input feat dim, DPAD: D padded to mult 32, H: hidden, KR: H padded to
// mult 32, NW: waves/block, IS_F32: fp32 input (layer 1), LAST: layer 4,
// DIR: 0 = forward, 1 = backward (time-reversed).
template <int D, int DPAD, int H, int KR, int NW, bool IS_F32, bool LAST, int DIR>
__global__ __launch_bounds__(NW * 32) void bilstm_dir(
    const void* __restrict__ xin, const float* __restrict__ Wk,
    const float* __restrict__ Wr, const float* __restrict__ bias,
    unsigned short* __restrict__ yout, float* __restrict__ ylast, int T) {
  constexpr int G    = 4 * H;
  constexpr int NT   = G / 16;        // N tiles of 16 gate columns
  constexpr int XC   = DPAD / 32;     // K chunks, input features
  constexpr int HC   = KR / 32;       // K chunks, recurrent features
  constexpr int NTH  = NW * 32;
  constexpr int NTPW = NT / NW;       // N tiles per wave
  static_assert(NT % NW == 0, "tile split must be exact");
  static_assert((16 * H) % NTH == 0, "elementwise split must be exact");
  constexpr int OUTF = 2 * H;
  constexpr int EW   = (16 * H) / NTH;            // elementwise iters
  constexpr bool ASY = (!IS_F32) && ASYNC_LDS_OK; // async-LDS staging path
  constexpr int NPF  = (16 * D + NTH - 1) / NTH;  // reg-prefetch elems/thread
  constexpr int NCP  = ASY ? ((8 * D) / NTH) : 1; // b32 async copies/thread
  static_assert(!ASY || (8 * D) % NTH == 0, "async copy split must be exact");

  // ---- LDS overlay: phase1 = weight staging, phase2 = per-step buffers ----
  constexpr int SXE = 16 * DPAD + 16;  // sX elems (+dummy slot for clamping)
  constexpr int NXB = ASY ? 2 : 1;     // double-buffered for async staging
  constexpr int P1  = G * DPAD * 2 + G * KR * 2;
  constexpr int P2  = NXB * SXE * 2 + 16 * KR * 2 + 16 * H * 4 + 16 * G * 4;
  __shared__ __align__(16) char smem[(P1 > P2) ? P1 : P2];
  unsigned short* sWkT = (unsigned short*)smem;                   // [4H][DPAD]
  unsigned short* sWrT = (unsigned short*)(smem + G * DPAD * 2);  // [4H][KR]
  unsigned short* sX0 = (unsigned short*)smem;                    // [NXB][16][DPAD]+pad
  unsigned short* sH = (unsigned short*)(smem + NXB * SXE * 2);   // [16][KR]
  float* sC = (float*)(smem + NXB * SXE * 2 + 16 * KR * 2);       // [16][H]
  float* sZ = (float*)(smem + NXB * SXE * 2 + 16 * KR * 2 + 16 * H * 4);  // [16][4H]

  const int tid  = threadIdx.x;
  const int lane = tid & 31;
  const int wave = __builtin_amdgcn_readfirstlane(tid >> 5);  // scalar wave id
  const int mrow = lane & 15;  // M row (A) == N col (B, C/D) for this lane
  const int half = lane >> 4;
  const int b0   = blockIdx.x * 16;

  // ---- phase 1: stage weights (bf16, transposed, zero K-padding) ----
  for (int i = tid; i < G * DPAD; i += NTH) {
    int n = i / DPAD, d = i - n * DPAD;
    sWkT[i] = (d < D) ? f2bf(Wk[(size_t)d * G + n]) : (unsigned short)0;
  }
  for (int i = tid; i < G * KR; i += NTH) {
    int n = i / KR, k = i - n * KR;
    sWrT[i] = (k < H) ? f2bf(Wr[(size_t)k * G + n]) : (unsigned short)0;
  }
  __syncthreads();

  // ---- hoist B fragments (32x16 bf16 layout) + bias into registers ----
  v8u bk[NTPW][XC], br[NTPW][HC];
  float bz[NTPW];
#pragma unroll
  for (int i = 0; i < NTPW; ++i) {
    const int nt = wave + i * NW;
    bz[i] = bias[nt * 16 + mrow];
    const unsigned short* wk = sWkT + (nt * 16 + mrow) * DPAD;
#pragma unroll
    for (int kc = 0; kc < XC; ++kc)
#pragma unroll
      for (int v = 0; v < 8; ++v)
        bk[i][kc][v] = *(const unsigned int*)(wk + kc * 32 + 8 * half + 2 * v + (v >= 4 ? 8 : 0));
    const unsigned short* wr = sWrT + (nt * 16 + mrow) * KR;
#pragma unroll
    for (int kc = 0; kc < HC; ++kc)
#pragma unroll
      for (int v = 0; v < 8; ++v)
        br[i][kc][v] = *(const unsigned int*)(wr + kc * 32 + 8 * half + 2 * v + (v >= 4 ? 8 : 0));
  }
  __syncthreads();  // weight staging area now dead -> reuse as phase-2 buffers

  // ---- staging bookkeeping (t-invariant, branch-free via dummy clamping) ----
  int prow[NPF];           // reg path: LDS short offset (dummy = 16*DPAD)
  size_t pbase[NPF];       // reg path: global short/float index base
  float pff[IS_F32 ? NPF : 1];
  unsigned short pfb[IS_F32 ? 1 : NPF];
  int ldsoff[NCP];         // async path: LDS short offset
  size_t gbase[NCP];       // async path: global short index base
  if constexpr (!ASY) {
#pragma unroll
    for (int p = 0; p < NPF; ++p) {
      int i = tid + p * NTH;
      bool act = i < 16 * D;
      int r = i / D, c = i - r * D;
      prow[p] = act ? (r * DPAD + c) : 16 * DPAD;            // dummy slot
      pbase[p] = act ? (((size_t)(b0 + r) * T) * D + c) : 0; // safe addr
    }
  } else {
    constexpr int HD = D / 2;  // dwords per row
#pragma unroll
    for (int p = 0; p < NCP; ++p) {
      int dw = tid + p * NTH;
      int r = dw / HD, cw = dw - r * HD;
      ldsoff[p] = r * DPAD + 2 * cw;
      gbase[p] = ((size_t)(b0 + r) * T) * D + 2 * cw;
    }
  }

  // ---- phase 2 init: zero buffers (pad cols must be 0), h, c ----
  for (int i = tid; i < NXB * SXE; i += NTH) sX0[i] = 0;
  for (int i = tid; i < 16 * KR; i += NTH) sH[i] = 0;
  for (int i = tid; i < 16 * H; i += NTH) sC[i] = 0.0f;
  __syncthreads();

  // ---- stage x_0 ----
  {
    const int tm0 = DIR ? (T - 1) : 0;
    if constexpr (ASY) {
#if ASYNC_LDS_OK
#pragma unroll
      for (int p = 0; p < NCP; ++p)
        async_g2l_b32((const unsigned short*)xin + gbase[p] + (size_t)tm0 * D,
                      sX0 + ldsoff[p]);
      wait_async0();
#endif
    } else {
#pragma unroll
      for (int p = 0; p < NPF; ++p) {
        if constexpr (IS_F32) pff[p] = ((const float*)xin)[pbase[p] + (size_t)tm0 * D];
        else pfb[p] = ((const unsigned short*)xin)[pbase[p] + (size_t)tm0 * D];
      }
#pragma unroll
      for (int p = 0; p < NPF; ++p)
        sX0[prow[p]] = IS_F32 ? f2bf(pff[p]) : pfb[p];
    }
  }
  __syncthreads();

  for (int t = 0; t < T; ++t) {
    const int tm = DIR ? (T - 1 - t) : t;
    const int tn = (t + 1 < T) ? (DIR ? (T - 2 - t) : (t + 1)) : 0;  // clamped
    const unsigned short* sX = sX0 + (ASY ? (t & 1) * SXE : 0);

    // issue async staging of x_{t+1} into the other buffer (fire-and-forget)
    if constexpr (ASY) {
#if ASYNC_LDS_OK
      unsigned short* dst = sX0 + ((t + 1) & 1) * SXE;
#pragma unroll
      for (int p = 0; p < NCP; ++p)
        async_g2l_b32((const unsigned short*)xin + gbase[p] + (size_t)tn * D,
                      dst + ldsoff[p]);
#endif
    }

    // build A fragments (16x32 bf16 layout, ISA 7.12.2)
    v8u ax[XC], ah[HC];
#pragma unroll
    for (int kc = 0; kc < XC; ++kc) {
      const unsigned short* row = &sX[mrow * DPAD + kc * 32];
#pragma unroll
      for (int v = 0; v < 4; ++v) {
        ax[kc][v]     = *(const unsigned int*)(row + 8 * half + 2 * v);
        ax[kc][v + 4] = *(const unsigned int*)(row + 16 + 8 * half + 2 * v);
      }
    }
#pragma unroll
    for (int kc = 0; kc < HC; ++kc) {
      const unsigned short* row = &sH[mrow * KR + kc * 32];
#pragma unroll
      for (int v = 0; v < 4; ++v) {
        ah[kc][v]     = *(const unsigned int*)(row + 8 * half + 2 * v);
        ah[kc][v + 4] = *(const unsigned int*)(row + 16 + 8 * half + 2 * v);
      }
    }

    // register-pipeline prefetch of x_{t+1} (f32 layer; hidden under compute)
    if constexpr (!ASY) {
#pragma unroll
      for (int p = 0; p < NPF; ++p) {
        if constexpr (IS_F32) pff[p] = ((const float*)xin)[pbase[p] + (size_t)tn * D];
        else pfb[p] = ((const unsigned short*)xin)[pbase[p] + (size_t)tn * D];
      }
    }

    // WMMA: z = x@Wk + h@Wr + b for this wave's N tiles (A and B in regs)
#pragma unroll
    for (int i = 0; i < NTPW; ++i) {
      const int nt = wave + i * NW;
      v8f acc = {0.f, 0.f, 0.f, 0.f, 0.f, 0.f, 0.f, 0.f};
#pragma unroll
      for (int kc = 0; kc < XC; ++kc)
        acc = __builtin_amdgcn_wmma_f32_16x16x32_bf16(
            false, __builtin_bit_cast(v16bf, ax[kc]), false,
            __builtin_bit_cast(v16bf, bk[i][kc]), (short)0, acc, false, false);
#pragma unroll
      for (int kc = 0; kc < HC; ++kc)
        acc = __builtin_amdgcn_wmma_f32_16x16x32_bf16(
            false, __builtin_bit_cast(v16bf, ah[kc]), false,
            __builtin_bit_cast(v16bf, br[i][kc]), (short)0, acc, false, false);
#pragma unroll
      for (int r = 0; r < 8; ++r)  // C/D layout: VGPR r -> M = r + 8*half
        sZ[(r + 8 * half) * G + nt * 16 + mrow] = acc[r] + bz[i];
    }
    __syncthreads();

    // elementwise LSTM cell update (gate order i,f,g,o)
#pragma unroll
    for (int q = 0; q < EW; ++q) {
      int idx = tid + q * NTH;
      int r = idx / H, j = idx - r * H;
      float zi = sZ[r * G + j];
      float zf = sZ[r * G + H + j];
      float zg = sZ[r * G + 2 * H + j];
      float zo = sZ[r * G + 3 * H + j];
      float c  = fsigm(zf) * sC[idx] + fsigm(zi) * ftanh(zg);
      sC[idx]  = c;
      float h  = fsigm(zo) * ftanh(c);
      sH[r * KR + j] = f2bf(h);
      if constexpr (!LAST) {
        yout[((size_t)(b0 + r) * T + tm) * OUTF + DIR * H + j] = f2bf(h);
      } else {
        if (t == T - 1) ylast[(size_t)(b0 + r) * (2 * H) + DIR * H + j] = h;
      }
    }
    // commit register-prefetched x_{t+1} (sX readers finished at barrier above)
    if constexpr (!ASY) {
#pragma unroll
      for (int p = 0; p < NPF; ++p)
        sX0[prow[p]] = IS_F32 ? f2bf(pff[p]) : pfb[p];
    } else {
#if ASYNC_LDS_OK
      wait_async0();  // this wave's copies done; barrier makes them visible
#endif
    }
    __syncthreads();
  }
}

__global__ void dense_sig(const float* __restrict__ h, const float* __restrict__ W,
                          const float* __restrict__ b, float* __restrict__ out, int B) {
  int i = blockIdx.x * blockDim.x + threadIdx.x;
  if (i >= B * 3) return;
  int bb = i / 3, k = i - bb * 3;
  float s = b[k];
#pragma unroll
  for (int j = 0; j < 8; ++j) s += h[bb * 8 + j] * W[j * 3 + k];
  out[i] = fsigm(s);
}

extern "C" void kernel_launch(void* const* d_in, const int* in_sizes, int n_in,
                              void* d_out, int out_size, void* d_ws, size_t ws_size,
                              hipStream_t stream) {
  (void)in_sizes; (void)n_in; (void)out_size; (void)ws_size;
  constexpr int B = 4096, T = 128;
  const float* x = (const float*)d_in[0];
  auto W = [&](int i) { return (const float*)d_in[i]; };

  // workspace: bf16 activation buffers per layer + final [B,8] f32 (~224 MB)
  unsigned short* y1 = (unsigned short*)d_ws;             // [B,T,128] bf16
  unsigned short* y2 = y1 + (size_t)B * T * 128;          // [B,T,64]  bf16
  unsigned short* y3 = y2 + (size_t)B * T * 64;           // [B,T,32]  bf16
  float* y4 = (float*)(y3 + (size_t)B * T * 32);          // [B,8]     f32

  dim3 grid(B / 16);
  // layer 1: D=78 (pad 96), H=64, f32 input (register pipeline)
  bilstm_dir<78, 96, 64, 64, 8, true, false, 0><<<grid, 256, 0, stream>>>(x, W(1), W(2), W(3), y1, nullptr, T);
  bilstm_dir<78, 96, 64, 64, 8, true, false, 1><<<grid, 256, 0, stream>>>(x, W(4), W(5), W(6), y1, nullptr, T);
  // layer 2: D=128, H=32 (async-LDS staging)
  bilstm_dir<128, 128, 32, 32, 8, false, false, 0><<<grid, 256, 0, stream>>>(y1, W(7), W(8), W(9), y2, nullptr, T);
  bilstm_dir<128, 128, 32, 32, 8, false, false, 1><<<grid, 256, 0, stream>>>(y1, W(10), W(11), W(12), y2, nullptr, T);
  // layer 3: D=64, H=16 (KR pad 32)
  bilstm_dir<64, 64, 16, 32, 4, false, false, 0><<<grid, 128, 0, stream>>>(y2, W(13), W(14), W(15), y3, nullptr, T);
  bilstm_dir<64, 64, 16, 32, 4, false, false, 1><<<grid, 128, 0, stream>>>(y2, W(16), W(17), W(18), y3, nullptr, T);
  // layer 4: D=32, H=4 (single 16x16 z tile, single wave), last state only
  bilstm_dir<32, 32, 4, 32, 1, false, true, 0><<<grid, 32, 0, stream>>>(y3, W(19), W(20), W(21), nullptr, y4, T);
  bilstm_dir<32, 32, 4, 32, 1, false, true, 1><<<grid, 32, 0, stream>>>(y3, W(22), W(23), W(24), nullptr, y4, T);
  // dense [8->3] + sigmoid
  dense_sig<<<(B * 3 + 255) / 256, 256, 0, stream>>>(y4, W(25), W(26), (float*)d_out, B);
}
